// Attention_63118839382659
// MI455X (gfx1250) — matrix operand order repaired and checked by reading, not compiled, not run on hardware
//
#include <hip/hip_runtime.h>

typedef __attribute__((ext_vector_type(16))) _Float16 v16h;
typedef __attribute__((ext_vector_type(8)))  _Float16 v8h;
typedef __attribute__((ext_vector_type(8)))  float    v8f;

#define BB   32
#define TT   512
#define DD   512
#define RR   32
#define NMOD 3
#define BTN  (BB*TT)          // 16384 rows
#define BETA 0.5f

// pack two f32 -> two f16 in one dword (for ds_store_b32 staging)
__device__ __forceinline__ unsigned int pack2h(float a, float b) {
    union { _Float16 h[2]; unsigned int u; } p;
    p.h[0] = (_Float16)a;
    p.h[1] = (_Float16)b;
    return p.u;
}

// ---------------------------------------------------------------------------
// Kernel 0a: pre-convert projection weights into f16, transposed, fragment-
// friendly layout: W16T[j][n][k],  n in [0,128) = 4 matrices x 32 cols,
// k in [0,512).  One-time, 384 KB output, L2-resident afterwards.
// ---------------------------------------------------------------------------
__global__ __launch_bounds__(256)
void convw_kernel(const float* __restrict__ Wq1, const float* __restrict__ Wq2,
                  const float* __restrict__ Wk1, const float* __restrict__ Wk2,
                  _Float16* __restrict__ W16T)
{
    const int id = blockIdx.x * 256 + threadIdx.x;   // 3*128*512 elements
    const int j  = id >> 16;                          // / (128*512)
    const int rem = id & 0xFFFF;
    const int n  = rem >> 9;                          // / 512
    const int k  = rem & 511;
    const int a  = n >> 5, r = n & 31;
    const float* Wp = (a == 0) ? Wq1 : (a == 1) ? Wq2 : (a == 2) ? Wk1 : Wk2;
    W16T[id] = (_Float16)Wp[((size_t)j * DD + k) * RR + r];
}

// ---------------------------------------------------------------------------
// Kernel 0b: Wa [j][k=1024][n=512] f32  ->  Wa16T [j][n=512][k=1024] f16,
// tiled 32x32 transpose through LDS so both global sides are coalesced.
// ---------------------------------------------------------------------------
__global__ __launch_bounds__(256)
void convwa_kernel(const float* __restrict__ Wa, _Float16* __restrict__ Wa16T)
{
    const int k0 = blockIdx.x * 32;     // 32 tiles in k
    const int n0 = blockIdx.y * 32;     // 16 tiles in n
    const int j  = blockIdx.z;
    const int tid = threadIdx.x;
    __shared__ _Float16 Tl[32][33];
#pragma unroll
    for (int e = 0; e < 4; ++e) {
        int idx = tid + e * 256;
        int kk = idx >> 5, nn = idx & 31;
        Tl[kk][nn] = (_Float16)Wa[((size_t)j * 1024 + k0 + kk) * DD + n0 + nn];
    }
    __syncthreads();
#pragma unroll
    for (int e = 0; e < 4; ++e) {
        int idx = tid + e * 256;
        int nn = idx >> 5, kk = idx & 31;
        Wa16T[((size_t)j * DD + n0 + nn) * 1024 + k0 + kk] = Tl[kk][nn];
    }
}

// ---------------------------------------------------------------------------
// Kernel 1: fused projections  Q = x_j @ [Wq1|Wq2|Wk1|Wk2] + b, then
//           u = q1*k1, v = q2*k2 elementwise, stored f16.
// Tile 128 rows x 128 cols, K=512 in chunks of 32; 8 waves, 16 rows each.
// A tile: x staged in LDS, 40-half row stride (80B = 5x16B -> aligned b128).
// B fragments: loaded directly from L2-resident W16T as contiguous v16h,
// software-pipelined one subtile ahead so loads ride over WMMAs.
// ---------------------------------------------------------------------------
__global__ __launch_bounds__(256)
void proj_uv_kernel(const float* __restrict__ x0, const float* __restrict__ x1,
                    const float* __restrict__ x2,
                    const _Float16* __restrict__ W16T,
                    const float* __restrict__ bq1, const float* __restrict__ bq2,
                    const float* __restrict__ bk1, const float* __restrict__ bk2,
                    _Float16* __restrict__ U, _Float16* __restrict__ V)
{
    const int j    = blockIdx.y;
    const float* x = (j == 0) ? x0 : (j == 1) ? x1 : x2;
    const int row0 = blockIdx.x * 128;
    const int tid  = threadIdx.x;
    const int lane = tid & 31;
    const int wave = tid >> 5;
    const int half = lane >> 4;
    const int rrow = lane & 15;

    __shared__ __align__(16) _Float16 As[128 * 40];        // 10 KB

    v8f zero = {};
    v8f acc[8];
#pragma unroll
    for (int i = 0; i < 8; ++i) acc[i] = zero;

    // loop-invariant base for this lane's B fragments (row n, contiguous k)
    const _Float16* wbaseT = &W16T[((size_t)j * 128 + rrow) * 512 + half * 16];

    for (int kc = 0; kc < 16; ++kc) {
        // ---- stage x chunk: 128x32, pair-packed (float2 load -> b32 LDS store)
#pragma unroll
        for (int e = 0; e < 8; ++e) {
            int idx = tid + e * 256;                 // 2048 pairs
            int m = idx >> 4, k2 = idx & 15;
            const float2 v2 = *(const float2*)&x[(size_t)(row0 + m) * DD + kc * 32 + 2 * k2];
            *(unsigned int*)&As[m * 40 + 2 * k2] = pack2h(v2.x, v2.y);
        }
        __syncthreads();

        // ---- A fragment: two aligned 16B LDS runs + shuffle
        const _Float16* rowp = &As[(wave * 16 + rrow) * 40];
        const v8h alo = *(const v8h*)(rowp + half * 16);
        const v8h ahi = *(const v8h*)(rowp + half * 16 + 8);
        const v16h af = __builtin_shufflevector(alo, ahi,
                            0, 1, 2, 3, 4, 5, 6, 7, 8, 9, 10, 11, 12, 13, 14, 15);

        // ---- B fragments pipelined one subtile ahead of their WMMA
        v16h bcur = *(const v16h*)(wbaseT + kc * 32);
#pragma unroll
        for (int nt = 0; nt < 8; ++nt) {
            v16h bnext;
            if (nt < 7)
                bnext = *(const v16h*)(wbaseT + (size_t)(nt + 1) * 16 * 512 + kc * 32);
            acc[nt] = __builtin_amdgcn_wmma_f32_16x16x32_f16(
                false, af, false, bcur, (short)0, acc[nt], false, false);
            bcur = bnext;
        }
        __syncthreads();
    }

    // epilogue: u = (q1+bq1)*(k1+bk1), v = (q2+bq2)*(k2+bk2)
    const int m0w = row0 + wave * 16;
#pragma unroll
    for (int p = 0; p < 2; ++p) {
        const int col = p * 16 + rrow;
        const float bq1v = bq1[j * RR + col], bq2v = bq2[j * RR + col];
        const float bk1v = bk1[j * RR + col], bk2v = bk2[j * RR + col];
#pragma unroll
        for (int vi = 0; vi < 8; ++vi) {
            const int row = m0w + vi + 8 * half;     // C layout: M = vgpr + 8*half
            const float uq = acc[p][vi]     + bq1v;
            const float uk = acc[4 + p][vi] + bk1v;
            const float vq = acc[2 + p][vi] + bq2v;
            const float vk = acc[6 + p][vi] + bk2v;
            U[((size_t)j * BTN + row) * RR + col] = (_Float16)(uq * uk);
            V[((size_t)j * BTN + row) * RR + col] = (_Float16)(vq * vk);
        }
    }
}

// ---------------------------------------------------------------------------
// Kernel 2: M[j][b] = (1/T) * U_b^T V_b   (32x32 per (j,b))  -- 100 MFLOP total
// ---------------------------------------------------------------------------
__global__ __launch_bounds__(256)
void moment_kernel(const _Float16* __restrict__ U, const _Float16* __restrict__ V,
                   float* __restrict__ M)
{
    const int j = blockIdx.x / BB;
    const int b = blockIdx.x % BB;
    const int tid = threadIdx.x;
    __shared__ _Float16 Us[64 * 33];
    __shared__ _Float16 Vs[64 * 33];
    float acc[4] = {0.f, 0.f, 0.f, 0.f};
    const size_t base = ((size_t)j * BTN + (size_t)b * TT) * RR;

    for (int tc = 0; tc < TT; tc += 64) {
#pragma unroll
        for (int e = 0; e < 8; ++e) {
            int idx = tid + e * 256;
            int t = idx >> 5, k = idx & 31;
            Us[t * 33 + k] = U[base + (size_t)(tc + t) * RR + k];
            Vs[t * 33 + k] = V[base + (size_t)(tc + t) * RR + k];
        }
        __syncthreads();
#pragma unroll
        for (int q = 0; q < 4; ++q) {
            int idx = tid + q * 256;
            int k = idx >> 5, l = idx & 31;
            float s = acc[q];
            for (int t = 0; t < 64; ++t)
                s += (float)Us[t * 33 + k] * (float)Vs[t * 33 + l];
            acc[q] = s;
        }
        __syncthreads();
    }
#pragma unroll
    for (int q = 0; q < 4; ++q) {
        int idx = tid + q * 256;
        M[((size_t)j * BB + b) * 1024 + idx] = acc[q] * (1.0f / TT);
    }
}

// ---------------------------------------------------------------------------
// Kernel 3: P[j][b] = M[l1][b] @ M[l2][b],  {l1,l2} = modules != j (ascending)
// ---------------------------------------------------------------------------
__global__ __launch_bounds__(256)
void chain_kernel(const float* __restrict__ M, float* __restrict__ P)
{
    const int j = blockIdx.x / BB;
    const int b = blockIdx.x % BB;
    const int l1 = (j == 0) ? 1 : 0;
    const int l2 = (j == 2) ? 1 : 2;
    const int tid = threadIdx.x;
    __shared__ float M1[32 * 33];
    __shared__ float M2[32 * 33];
#pragma unroll
    for (int e = 0; e < 4; ++e) {
        int idx = tid + e * 256;
        int r = idx >> 5, c = idx & 31;
        M1[r * 33 + c] = M[((size_t)l1 * BB + b) * 1024 + idx];
        M2[r * 33 + c] = M[((size_t)l2 * BB + b) * 1024 + idx];
    }
    __syncthreads();
#pragma unroll
    for (int q = 0; q < 4; ++q) {
        int idx = tid + q * 256;
        int k = idx >> 5, o = idx & 31;
        float s = 0.f;
#pragma unroll
        for (int l = 0; l < 32; ++l) s += M1[k * 33 + l] * M2[l * 33 + o];
        P[((size_t)j * BB + b) * 1024 + idx] = s;
    }
}

// ---------------------------------------------------------------------------
// Kernel 4: dominant GEMM.  out = (vec(u ⊗ (v·P)) @ Wa_j + ba_j) * x + β x
// Block: 32 rows x 256 cols; K=1024 in 32 chunks of 32 (chunk c <-> outer k=c:
// A rows are u[r][c] * w[r][0..31], synthesized in registers from preloaded w).
// B fragments and u scalars software-pipelined one K-chunk ahead; no LDS
// staging and no barriers in the K-loop.
// ---------------------------------------------------------------------------
__global__ __launch_bounds__(256)
void out_kernel(const float* __restrict__ x0, const float* __restrict__ x1,
                const float* __restrict__ x2,
                const _Float16* __restrict__ U, const _Float16* __restrict__ V,
                const float* __restrict__ P,  const _Float16* __restrict__ Wa16T,
                const float* __restrict__ ba, float* __restrict__ out)
{
    const int j    = blockIdx.z;
    const float* x = (j == 0) ? x0 : (j == 1) ? x1 : x2;
    const int row0  = blockIdx.x * 32;         // 32 rows, all in one batch b
    const int nbase = blockIdx.y * 256;
    const int b     = row0 / TT;
    const int tid   = threadIdx.x;
    const int lane  = tid & 31;
    const int wave  = tid >> 5;
    const int half  = lane >> 4;
    const int rrow  = lane & 15;

    __shared__ float Ps[32 * 33];                          // 4.2 KB
    __shared__ float us[32 * 32];                          // 4 KB
    __shared__ float vs[32 * 32];                          // 4 KB
    __shared__ float wls[32 * 32];                         // 4 KB

    // ---- stage P, u, v
#pragma unroll
    for (int e = 0; e < 4; ++e) {
        int idx = tid + e * 256;
        Ps[(idx >> 5) * 33 + (idx & 31)] = P[((size_t)j * BB + b) * 1024 + idx];
    }
#pragma unroll
    for (int e = 0; e < 4; ++e) {
        int idx = tid + e * 256;
        int m = idx >> 5, c = idx & 31;
        us[m * 32 + c] = (float)U[((size_t)j * BTN + row0 + m) * RR + c];
        vs[m * 32 + c] = (float)V[((size_t)j * BTN + row0 + m) * RR + c];
    }
    __syncthreads();

    // ---- w = v @ P   (32x32)
#pragma unroll
    for (int e = 0; e < 4; ++e) {
        int idx = tid + e * 256;
        int m = idx >> 5, o = idx & 31;
        float s = 0.f;
#pragma unroll
        for (int l = 0; l < 32; ++l) s += vs[m * 32 + l] * Ps[l * 33 + o];
        wls[m * 32 + o] = s;
    }
    __syncthreads();

    // ---- preload loop-invariant w fragment rows into registers
    float wreg[2][16];
#pragma unroll
    for (int ms = 0; ms < 2; ++ms) {
        const float* wr = &wls[(ms * 16 + rrow) * 32];
#pragma unroll
        for (int i = 0; i < 8; ++i) {
            wreg[ms][i]     = wr[half * 16 + i];
            wreg[ms][8 + i] = wr[half * 16 + 8 + i];
        }
    }

    v8f zero = {};
    v8f acc[2][2];
    acc[0][0] = zero; acc[0][1] = zero; acc[1][0] = zero; acc[1][1] = zero;

    // per-lane B fragment bases for the two N-subtiles owned by this wave
    const _Float16* bT0 = &Wa16T[((size_t)j * DD + nbase + (wave * 2 + 0) * 16 + rrow) * 1024
                                 + half * 16];
    const _Float16* bT1 = &Wa16T[((size_t)j * DD + nbase + (wave * 2 + 1) * 16 + rrow) * 1024
                                 + half * 16];

    // ---- software pipeline: B fragments + u scalars one chunk ahead
    v16h b0 = *(const v16h*)(bT0);
    v16h b1 = *(const v16h*)(bT1);
    float uv0 = us[(0 * 16 + rrow) * 32 + 0];
    float uv1 = us[(1 * 16 + rrow) * 32 + 0];

#pragma unroll 4
    for (int c = 0; c < 32; ++c) {
        v16h nb0, nb1;
        float nuv0, nuv1;
        if (c < 31) {
            nb0  = *(const v16h*)(bT0 + (c + 1) * 32);
            nb1  = *(const v16h*)(bT1 + (c + 1) * 32);
            nuv0 = us[(0 * 16 + rrow) * 32 + c + 1];
            nuv1 = us[(1 * 16 + rrow) * 32 + c + 1];
        }
        // A fragments synthesized in registers: f[r, k] = u[r][c] * w[r][k]
        v16h af0, af1;
#pragma unroll
        for (int i = 0; i < 16; ++i) af0[i] = (_Float16)(uv0 * wreg[0][i]);
#pragma unroll
        for (int i = 0; i < 16; ++i) af1[i] = (_Float16)(uv1 * wreg[1][i]);

        acc[0][0] = __builtin_amdgcn_wmma_f32_16x16x32_f16(
            false, af0, false, b0, (short)0, acc[0][0], false, false);
        acc[1][0] = __builtin_amdgcn_wmma_f32_16x16x32_f16(
            false, af1, false, b0, (short)0, acc[1][0], false, false);
        acc[0][1] = __builtin_amdgcn_wmma_f32_16x16x32_f16(
            false, af0, false, b1, (short)0, acc[0][1], false, false);
        acc[1][1] = __builtin_amdgcn_wmma_f32_16x16x32_f16(
            false, af1, false, b1, (short)0, acc[1][1], false, false);

        b0 = nb0; b1 = nb1; uv0 = nuv0; uv1 = nuv1;
    }

    // ---- fused epilogue: out = (acc + ba) * x + beta * x
#pragma unroll
    for (int t = 0; t < 2; ++t) {
        const int n = nbase + (wave * 2 + t) * 16 + rrow;
        const float bav = ba[j * DD + n];
#pragma unroll
        for (int ms = 0; ms < 2; ++ms) {
#pragma unroll
            for (int vi = 0; vi < 8; ++vi) {
                const int row = row0 + ms * 16 + vi + 8 * half;
                const float xv = x[(size_t)row * DD + n];
                const float a_ = acc[ms][t][vi] + bav;
                out[(size_t)j * BTN * DD + (size_t)row * DD + n] = a_ * xv + BETA * xv;
            }
        }
    }
}

// ---------------------------------------------------------------------------
extern "C" void kernel_launch(void* const* d_in, const int* in_sizes, int n_in,
                              void* d_out, int out_size, void* d_ws, size_t ws_size,
                              hipStream_t stream)
{
    (void)in_sizes; (void)n_in; (void)out_size; (void)ws_size;

    const float* x0  = (const float*)d_in[0];
    const float* x1  = (const float*)d_in[1];
    const float* x2  = (const float*)d_in[2];
    const float* Wq1 = (const float*)d_in[3];
    const float* bq1 = (const float*)d_in[4];
    const float* Wq2 = (const float*)d_in[5];
    const float* bq2 = (const float*)d_in[6];
    const float* Wk1 = (const float*)d_in[7];
    const float* bk1 = (const float*)d_in[8];
    const float* Wk2 = (const float*)d_in[9];
    const float* bk2 = (const float*)d_in[10];
    const float* Wa  = (const float*)d_in[11];
    const float* ba  = (const float*)d_in[12];
    float* out = (float*)d_out;

    // workspace layout: U[f16] 3MB, V[f16] 3MB, M[f32] .38MB, P[f32] .38MB,
    //                   W16T[f16] .38MB, Wa16T[f16] 3MB   (~10.2 MB total)
    _Float16* Uw    = (_Float16*)d_ws;
    _Float16* Vw    = Uw + (size_t)NMOD * BTN * RR;
    float*    Mw    = (float*)(Vw + (size_t)NMOD * BTN * RR);
    float*    Pw    = Mw + (size_t)NMOD * BB * RR * RR;
    _Float16* W16T  = (_Float16*)(Pw + (size_t)NMOD * BB * RR * RR);
    _Float16* Wa16T = W16T + (size_t)NMOD * 128 * DD;

    // one-time weight conversions (L2-resident afterwards)
    convw_kernel<<<dim3((NMOD * 128 * DD) / 256), 256, 0, stream>>>(
        Wq1, Wq2, Wk1, Wk2, W16T);
    convwa_kernel<<<dim3(32, 16, NMOD), 256, 0, stream>>>(Wa, Wa16T);

    dim3 g1(BTN / 128, NMOD);
    proj_uv_kernel<<<g1, 256, 0, stream>>>(x0, x1, x2, W16T,
                                           bq1, bq2, bk1, bk2, Uw, Vw);

    moment_kernel<<<dim3(NMOD * BB), 256, 0, stream>>>(Uw, Vw, Mw);

    chain_kernel<<<dim3(NMOD * BB), 256, 0, stream>>>(Mw, Pw);

    dim3 g4(BTN / 32, DD / 256, NMOD);
    out_kernel<<<g4, 256, 0, stream>>>(x0, x1, x2, Uw, Vw, Pw, Wa16T, ba, out);
}
